// SpikeDrivenSelfAttention_27315992003113
// MI455X (gfx1250) — 16-bit and 8-bit matrix operand order repaired and checked by reading, compiled, not run on hardware
//
#include <hip/hip_runtime.h>
#include <hip/hip_bf16.h>

typedef __attribute__((ext_vector_type(16))) _Float16 v16h;
typedef __attribute__((ext_vector_type(8)))  _Float16 v8h;
typedef __attribute__((ext_vector_type(8)))  float    v8f;
typedef __attribute__((ext_vector_type(8)))  int      v8i;

#define B_  2
#define T_  2048
#define C_  1024
#define H_  16
#define D_  64

__global__ void cvt_f32_to_f16(const float* __restrict__ src,
                               _Float16* __restrict__ dst, int n) {
  int i = blockIdx.x * blockDim.x + threadIdx.x;
  if (i < n) dst[i] = (_Float16)src[i];
}

__device__ __forceinline__ v16h cat8(v8h lo, v8h hi) {
  v16h r;
#pragma unroll
  for (int i = 0; i < 8; ++i) { r[i] = lo[i]; r[i + 8] = hi[i]; }
  return r;
}

__device__ __forceinline__ v8f wmma_f16(v16h a, v16h b, v8f c) {
  return __builtin_amdgcn_wmma_f32_16x16x32_f16(false, a, false, b,
                                                (short)0, c, false, false);
}

__global__ void gemm_qkv(const _Float16* __restrict__ xh,
                         const _Float16* __restrict__ wq,
                         const _Float16* __restrict__ wk,
                         const _Float16* __restrict__ wv,
                         const float* __restrict__ bq,
                         const float* __restrict__ bk,
                         const float* __restrict__ bv,
                         unsigned char* __restrict__ qsb,
                         unsigned char* __restrict__ ksb,
                         _Float16* __restrict__ vt) {
  const int which = blockIdx.y;
  const _Float16* W   = (which == 0) ? wq : (which == 1) ? wk : wv;
  const float*   bias = (which == 0) ? bq : (which == 1) ? bk : bv;

  const int wave = threadIdx.x >> 5, lane = threadIdx.x & 31;
  const int hi = lane >> 4, ln = lane & 15;
  const int tile = blockIdx.x * 8 + wave;
  const int mt = tile >> 5, nt = tile & 31;
  const _Float16* arow0 = xh + (size_t)(mt * 32 + ln) * C_;
  const _Float16* arow1 = arow0 + (size_t)16 * C_;
  const int bn0 = nt * 32 + ln, bn1 = bn0 + 16;
  const _Float16* brow0 = W + (size_t)bn0 * C_;
  const _Float16* brow1 = W + (size_t)bn1 * C_;

  v8f acc00 = {}, acc01 = {}, acc10 = {}, acc11 = {};
#pragma unroll 2
  for (int k0 = 0; k0 < C_; k0 += 32) {
    const int ka = k0 + (hi ? 8 : 0);
    v16h A0 = cat8(*(const v8h*)(arow0 + ka), *(const v8h*)(arow0 + ka + 16));
    v16h A1 = cat8(*(const v8h*)(arow1 + ka), *(const v8h*)(arow1 + ka + 16));
    const int kb = k0 + (hi ? 8 : 0);
    v16h B0 = cat8(*(const v8h*)(brow0 + kb), *(const v8h*)(brow0 + kb + 16));
    v16h B1 = cat8(*(const v8h*)(brow1 + kb), *(const v8h*)(brow1 + kb + 16));
    acc00 = wmma_f16(A0, B0, acc00);
    acc01 = wmma_f16(A0, B1, acc01);
    acc10 = wmma_f16(A1, B0, acc10);
    acc11 = wmma_f16(A1, B1, acc11);
  }

#pragma unroll
  for (int mi = 0; mi < 2; ++mi) {
#pragma unroll
    for (int ni = 0; ni < 2; ++ni) {
      v8f acc = (mi == 0) ? ((ni == 0) ? acc00 : acc01)
                          : ((ni == 0) ? acc10 : acc11);
      const int bn = (ni == 0) ? bn0 : bn1;
      const float bcol = bias[bn];
      const int h = bn >> 6, d = bn & 63;
      const int row0 = mt * 32 + mi * 16 + 8 * hi;
      if (which < 2) {
        unsigned char* out = (which == 0) ? qsb : ksb;
#pragma unroll
        for (int r = 0; r < 8; ++r) {
          int row = row0 + r;
          int bb = row >> 11, t = row & (T_ - 1);
          float val = acc[r] + bcol;
          out[(((size_t)(bb * H_ + h) * T_ + t) << 6) + d] =
              (val > 1.0f) ? (unsigned char)1 : (unsigned char)0;
        }
      } else {
        int bb = row0 >> 11, t0 = row0 & (T_ - 1);
        v8h hv;
#pragma unroll
        for (int r = 0; r < 8; ++r) hv[r] = (_Float16)(acc[r] + bcol);
        *(v8h*)(vt + ((size_t)(bb * H_ + h) * D_ + d) * T_ + t0) = hv;
      }
    }
  }
}

__global__ void attn_kernel(const unsigned char* __restrict__ qsb,
                            const unsigned char* __restrict__ ksb,
                            const _Float16* __restrict__ vt,
                            _Float16* __restrict__ att) {
  __shared__ __align__(16) _Float16 pP[8][2][16][40];

  const int bh = blockIdx.x >> 3;
  const int chunk = blockIdx.x & 7;
  const int wave = threadIdx.x >> 5, lane = threadIdx.x & 31;
  const int hi = lane >> 4, ln = lane & 15;
  const int qbase = chunk * 256 + wave * 32;

  const unsigned char* qrow0 = qsb + ((size_t)bh * T_ + qbase + ln) * D_;
  const unsigned char* qrow1 = qrow0 + (size_t)16 * D_;
  v8i a8_0, a8_1;
#pragma unroll
  for (int g = 0; g < 4; ++g) {
    int off = (g >> 1) * 32 + (g & 1) * 8 + (hi ? 16 : 0);
    int2 p0 = *(const int2*)(qrow0 + off);
    int2 p1 = *(const int2*)(qrow1 + off);
    a8_0[2 * g] = p0.x; a8_0[2 * g + 1] = p0.y;
    a8_1[2 * g] = p1.x; a8_1[2 * g + 1] = p1.y;
  }

  v8f acc0[4], acc1[4];
#pragma unroll
  for (int dt = 0; dt < 4; ++dt) { acc0[dt] = {}; acc1[dt] = {}; }
  float dsum0[8], dsum1[8];
#pragma unroll
  for (int r = 0; r < 8; ++r) { dsum0[r] = 0.0f; dsum1[r] = 0.0f; }

  const int kb8 = hi ? 16 : 0;

  for (int j = 0; j < T_; j += 32) {
#pragma unroll
    for (int jj = 0; jj < 2; ++jj) {
      const unsigned char* krow =
          ksb + ((size_t)bh * T_ + j + jj * 16 + ln) * D_ + kb8;
      int4 c0 = *(const int4*)(krow);
      int4 c1 = *(const int4*)(krow + 32);
      v8i b8;
      b8[0] = c0.x; b8[1] = c0.y; b8[2] = c0.z; b8[3] = c0.w;
      b8[4] = c1.x; b8[5] = c1.y; b8[6] = c1.z; b8[7] = c1.w;
      v8i zero = {};
      v8i sc0 = __builtin_amdgcn_wmma_i32_16x16x64_iu8(false, a8_0, false, b8,
                                                       zero, false, false);
      v8i sc1 = __builtin_amdgcn_wmma_i32_16x16x64_iu8(false, a8_1, false, b8,
                                                       zero, false, false);
#pragma unroll
      for (int r = 0; r < 8; ++r) {
        float e0 = __expf((float)sc0[r] * 0.125f);
        float e1 = __expf((float)sc1[r] * 0.125f);
        dsum0[r] += e0;
        dsum1[r] += e1;
        pP[wave][0][r + 8 * hi][jj * 16 + ln] = (_Float16)e0;
        pP[wave][1][r + 8 * hi][jj * 16 + ln] = (_Float16)e1;
      }
    }
    const int ka = hi ? 8 : 0;
    v16h pa0 = cat8(*(const v8h*)&pP[wave][0][ln][ka],
                    *(const v8h*)&pP[wave][0][ln][ka + 16]);
    v16h pa1 = cat8(*(const v8h*)&pP[wave][1][ln][ka],
                    *(const v8h*)&pP[wave][1][ln][ka + 16]);

    const _Float16* vbase = vt + (size_t)bh * D_ * T_ + j + (hi ? 8 : 0);
#pragma unroll
    for (int dt = 0; dt < 4; ++dt) {
      v16h Bv = cat8(*(const v8h*)(vbase + (size_t)(dt * 16 + ln) * T_),
                     *(const v8h*)(vbase + (size_t)(dt * 16 + ln) * T_ + 16));
      acc0[dt] = wmma_f16(pa0, Bv, acc0[dt]);
      acc1[dt] = wmma_f16(pa1, Bv, acc1[dt]);
    }
  }

#pragma unroll
  for (int r = 0; r < 8; ++r) {
    dsum0[r] += __shfl_xor(dsum0[r], 1, 16);
    dsum0[r] += __shfl_xor(dsum0[r], 2, 16);
    dsum0[r] += __shfl_xor(dsum0[r], 4, 16);
    dsum0[r] += __shfl_xor(dsum0[r], 8, 16);
    dsum1[r] += __shfl_xor(dsum1[r], 1, 16);
    dsum1[r] += __shfl_xor(dsum1[r], 2, 16);
    dsum1[r] += __shfl_xor(dsum1[r], 4, 16);
    dsum1[r] += __shfl_xor(dsum1[r], 8, 16);
  }

  const int b = bh >> 4, h = bh & 15;
#pragma unroll
  for (int ti = 0; ti < 2; ++ti) {
#pragma unroll
    for (int r = 0; r < 8; ++r) {
      float inv = 1.0f / ((ti == 0) ? dsum0[r] : dsum1[r]);
      int t = qbase + ti * 16 + 8 * hi + r;
      _Float16* orow = att + ((size_t)(b * T_ + t)) * C_ + h * D_ + ln;
      if (ti == 0) {
        orow[0]  = (_Float16)(acc0[0][r] * inv);
        orow[16] = (_Float16)(acc0[1][r] * inv);
        orow[32] = (_Float16)(acc0[2][r] * inv);
        orow[48] = (_Float16)(acc0[3][r] * inv);
      } else {
        orow[0]  = (_Float16)(acc1[0][r] * inv);
        orow[16] = (_Float16)(acc1[1][r] * inv);
        orow[32] = (_Float16)(acc1[2][r] * inv);
        orow[48] = (_Float16)(acc1[3][r] * inv);
      }
    }
  }
}

__global__ void gemm_proj(const _Float16* __restrict__ ah,
                          const _Float16* __restrict__ wo,
                          const float* __restrict__ bo,
                          float* __restrict__ out) {
  const int wave = threadIdx.x >> 5, lane = threadIdx.x & 31;
  const int hi = lane >> 4, ln = lane & 15;
  const int tile = blockIdx.x * 8 + wave;
  const int mt = tile >> 5, nt = tile & 31;
  const _Float16* arow0 = ah + (size_t)(mt * 32 + ln) * C_;
  const _Float16* arow1 = arow0 + (size_t)16 * C_;
  const int bn0 = nt * 32 + ln, bn1 = bn0 + 16;
  const _Float16* brow0 = wo + (size_t)bn0 * C_;
  const _Float16* brow1 = wo + (size_t)bn1 * C_;

  v8f acc00 = {}, acc01 = {}, acc10 = {}, acc11 = {};
#pragma unroll 2
  for (int k0 = 0; k0 < C_; k0 += 32) {
    const int ka = k0 + (hi ? 8 : 0);
    v16h A0 = cat8(*(const v8h*)(arow0 + ka), *(const v8h*)(arow0 + ka + 16));
    v16h A1 = cat8(*(const v8h*)(arow1 + ka), *(const v8h*)(arow1 + ka + 16));
    const int kb = k0 + (hi ? 8 : 0);
    v16h B0 = cat8(*(const v8h*)(brow0 + kb), *(const v8h*)(brow0 + kb + 16));
    v16h B1 = cat8(*(const v8h*)(brow1 + kb), *(const v8h*)(brow1 + kb + 16));
    acc00 = wmma_f16(A0, B0, acc00);
    acc01 = wmma_f16(A0, B1, acc01);
    acc10 = wmma_f16(A1, B0, acc10);
    acc11 = wmma_f16(A1, B1, acc11);
  }

  const float bc0 = bo[bn0], bc1 = bo[bn1];
#pragma unroll
  for (int mi = 0; mi < 2; ++mi) {
    const int row0 = mt * 32 + mi * 16 + 8 * hi;
#pragma unroll
    for (int r = 0; r < 8; ++r) {
      const size_t rowoff = (size_t)(row0 + r) * C_;
      out[rowoff + bn0] = ((mi == 0) ? acc00[r] : acc10[r]) + bc0;
      out[rowoff + bn1] = ((mi == 0) ? acc01[r] : acc11[r]) + bc1;
    }
  }
}

extern "C" void kernel_launch(void* const* d_in, const int* in_sizes, int n_in,
                              void* d_out, int out_size, void* d_ws, size_t ws_size,
                              hipStream_t stream) {
  const float* x  = (const float*)d_in[0];
  const float* Wq = (const float*)d_in[1];
  const float* bq = (const float*)d_in[2];
  const float* Wk = (const float*)d_in[3];
  const float* bk = (const float*)d_in[4];
  const float* Wv = (const float*)d_in[5];
  const float* bv = (const float*)d_in[6];
  const float* Wo = (const float*)d_in[7];
  const float* bo = (const float*)d_in[8];

  char* ws = (char*)d_ws;
  const size_t MB = (size_t)1 << 20;
  _Float16*      xh  = (_Float16*)(ws + 0);
  _Float16*      wqh = (_Float16*)(ws + 8  * MB);
  _Float16*      wkh = (_Float16*)(ws + 10 * MB);
  _Float16*      wvh = (_Float16*)(ws + 12 * MB);
  _Float16*      woh = (_Float16*)(ws + 14 * MB);
  unsigned char* qsb = (unsigned char*)(ws + 16 * MB);
  unsigned char* ksb = (unsigned char*)(ws + 20 * MB);
  _Float16*      vt  = (_Float16*)(ws + 24 * MB);
  _Float16*      att = (_Float16*)(ws + 32 * MB);

  const int NX = B_ * T_ * C_;
  const int NW = C_ * C_;
  cvt_f32_to_f16<<<(NX + 255) / 256, 256, 0, stream>>>(x,  xh,  NX);
  cvt_f32_to_f16<<<(NW + 255) / 256, 256, 0, stream>>>(Wq, wqh, NW);
  cvt_f32_to_f16<<<(NW + 255) / 256, 256, 0, stream>>>(Wk, wkh, NW);
  cvt_f32_to_f16<<<(NW + 255) / 256, 256, 0, stream>>>(Wv, wvh, NW);
  cvt_f32_to_f16<<<(NW + 255) / 256, 256, 0, stream>>>(Wo, woh, NW);

  gemm_qkv<<<dim3(512, 3), 256, 0, stream>>>(xh, wqh, wkh, wvh,
                                             bq, bk, bv, qsb, ksb, vt);
  attn_kernel<<<256, 256, 0, stream>>>(qsb, ksb, vt, att);

  gemm_proj<<<512, 256, 0, stream>>>(att, woh, bo, (float*)d_out);
}
